// GCNEvaluator_28870770164391
// MI455X (gfx1250) — matrix operand order repaired and checked by reading, not compiled, not run on hardware
//
#include <hip/hip_runtime.h>

#define N_NODES 100000
#define N_EDGES 1600000
#define IN_DIM  128
#define EDGE_DIM 8
#define HID     64
#define N_LAYERS 3

typedef __attribute__((ext_vector_type(16))) _Float16 v16h;
typedef __attribute__((ext_vector_type(8)))  float    v8f;

static __device__ __forceinline__ v8f wmma_f16(v16h a, v16h b, v8f c) {
  // D = A(16x32 f16) x B(32x16 f16) + C(16x16 f32)
  return __builtin_amdgcn_wmma_f32_16x16x32_f16(false, a, false, b, (short)0, c,
                                                false, false);
}

// ---------------------------------------------------------------------------
// Edge MLP: ew = relu(relu(edge_attr @ W1^T) @ W2^T)   [E,8] -> [E,64]
// Grid-stride over 16-edge tiles; weight B-fragments hoisted to registers.
// W1 staged pre-padded (K 8->32) so B loads are unconditional aligned b128s.
// ---------------------------------------------------------------------------
__global__ __launch_bounds__(256) void edge_mlp_kernel(
    const float* __restrict__ edge_attr, const float* __restrict__ W1,
    const float* __restrict__ W2, float* __restrict__ ew)
{
  __shared__ alignas(32) _Float16 sB1[HID * 32];   // [n][k] K zero-padded to 32
  __shared__ alignas(32) _Float16 sW2[HID * HID];  // [n][k]
  __shared__ alignas(32) _Float16 sH[8][16][HID];  // per-wave 16x64 intermediate

  int tid = threadIdx.x;
  for (int i = tid; i < HID * 32; i += 256) {
    int n = i >> 5, k = i & 31;
    sB1[i] = (k < EDGE_DIM) ? (_Float16)W1[n * EDGE_DIM + k] : (_Float16)0.f;
  }
  for (int i = tid; i < HID * HID; i += 256) sW2[i] = (_Float16)W2[i];
  __syncthreads();

  int wave = tid >> 5, lane = tid & 31;
  int half = lane >> 4, ln = lane & 15;

  // hoist loop-invariant B fragments
  v16h b1[4], b2[2][4];
#pragma unroll
  for (int nt = 0; nt < 4; ++nt) {
    int n = nt * 16 + ln;
    b1[nt] = *reinterpret_cast<const v16h*>(sB1 + n * 32 + 16 * half);
#pragma unroll
    for (int kc = 0; kc < 2; ++kc)
      b2[kc][nt] = *reinterpret_cast<const v16h*>(sW2 + n * HID + kc * 32 + 16 * half);
  }

  const long long nTiles = N_EDGES / 16;
  const long long stride = (long long)gridDim.x * 8;
  for (long long tile = (long long)blockIdx.x * 8 + wave; tile < nTiles;
       tile += stride) {
    long long e0 = tile * 16;

    // A = edge_attr rows (K=8 live, rest zero)
    v16h a1 = {};
    if (half == 0) {
      const float* ea = edge_attr + (e0 + ln) * EDGE_DIM;
#pragma unroll
      for (int j = 0; j < 8; ++j) a1[j] = (_Float16)ea[j];
    }
    v8f h1[4];
#pragma unroll
    for (int nt = 0; nt < 4; ++nt) {
      v8f c = {};
      h1[nt] = wmma_f16(a1, b1[nt], c);
    }
#pragma unroll
    for (int nt = 0; nt < 4; ++nt)
#pragma unroll
      for (int r = 0; r < 8; ++r) {
        float v = h1[nt][r];
        sH[wave][r + 8 * half][nt * 16 + ln] = (_Float16)(v > 0.f ? v : 0.f);
      }

    v8f c2[4] = {};
#pragma unroll
    for (int kc = 0; kc < 2; ++kc) {
      int k0 = kc * 32;
      v16h a;
#pragma unroll
      for (int j = 0; j < 8; ++j) {
        a[j]     = sH[wave][ln][k0 + 16 * half + j];
        a[8 + j] = sH[wave][ln][k0 + 16 * half + 8 + j];
      }
#pragma unroll
      for (int nt = 0; nt < 4; ++nt) c2[nt] = wmma_f16(a, b2[kc][nt], c2[nt]);
    }
#pragma unroll
    for (int nt = 0; nt < 4; ++nt)
#pragma unroll
      for (int r = 0; r < 8; ++r) {
        float v = c2[nt][r];
        ew[(e0 + r + 8 * half) * HID + nt * 16 + ln] = v > 0.f ? v : 0.f;
      }
  }
}

// ---------------------------------------------------------------------------
// i_lin: x_ = x @ Wi^T + bi   [N,128] -> [N,64]; bias preloaded into C;
// grid-stride over 16-node tiles with register-resident B fragments.
// ---------------------------------------------------------------------------
__global__ __launch_bounds__(256) void ilin_kernel(
    const float* __restrict__ x, const float* __restrict__ Wi,
    const float* __restrict__ bi, float* __restrict__ xh)
{
  __shared__ alignas(32) _Float16 sW[HID * IN_DIM];
  int tid = threadIdx.x;
  for (int i = tid; i < HID * IN_DIM; i += 256) sW[i] = (_Float16)Wi[i];
  __syncthreads();

  int wave = tid >> 5, lane = tid & 31;
  int half = lane >> 4, ln = lane & 15;

  v16h b[4][4];
  float bv[4];
#pragma unroll
  for (int nt = 0; nt < 4; ++nt) {
    int n = nt * 16 + ln;
    bv[nt] = bi[n];
#pragma unroll
    for (int kc = 0; kc < 4; ++kc)
      b[kc][nt] = *reinterpret_cast<const v16h*>(sW + n * IN_DIM + kc * 32 + 16 * half);
  }

  const long long nTiles = (N_NODES + 15) / 16;
  const long long stride = (long long)gridDim.x * 8;
  for (long long tile = (long long)blockIdx.x * 8 + wave; tile < nTiles;
       tile += stride) {
    long long n0 = tile * 16;
    v8f acc[4];
#pragma unroll
    for (int nt = 0; nt < 4; ++nt)
#pragma unroll
      for (int r = 0; r < 8; ++r) acc[nt][r] = bv[nt];

    const float* xr = x + (n0 + ln) * IN_DIM;
#pragma unroll
    for (int kc = 0; kc < 4; ++kc) {
      int k0 = kc * 32;
      v16h a;
#pragma unroll
      for (int j = 0; j < 8; ++j) {
        a[j]     = (_Float16)xr[k0 + 16 * half + j];
        a[8 + j] = (_Float16)xr[k0 + 16 * half + 8 + j];
      }
#pragma unroll
      for (int nt = 0; nt < 4; ++nt) acc[nt] = wmma_f16(a, b[kc][nt], acc[nt]);
    }
#pragma unroll
    for (int nt = 0; nt < 4; ++nt)
#pragma unroll
      for (int r = 0; r < 8; ++r)
        xh[(n0 + r + 8 * half) * HID + nt * 16 + ln] = acc[nt][r];
  }
}

// ---------------------------------------------------------------------------
// mix: h = concat(x_, agg) @ Wl^T + x_   (residual preloaded into C)
// ---------------------------------------------------------------------------
__global__ __launch_bounds__(256) void mix_kernel(
    const float* __restrict__ xh, const float* __restrict__ agg,
    const float* __restrict__ Wl, float* __restrict__ hout)
{
  __shared__ alignas(32) _Float16 sW[HID * 2 * HID];
  int tid = threadIdx.x;
  for (int i = tid; i < HID * 2 * HID; i += 256) sW[i] = (_Float16)Wl[i];
  __syncthreads();

  int wave = tid >> 5, lane = tid & 31;
  int half = lane >> 4, ln = lane & 15;

  v16h b[4][4];
#pragma unroll
  for (int nt = 0; nt < 4; ++nt) {
    int n = nt * 16 + ln;
#pragma unroll
    for (int kc = 0; kc < 4; ++kc)
      b[kc][nt] = *reinterpret_cast<const v16h*>(sW + n * 2 * HID + kc * 32 + 16 * half);
  }

  const long long nTiles = (N_NODES + 15) / 16;
  const long long stride = (long long)gridDim.x * 8;
  for (long long tile = (long long)blockIdx.x * 8 + wave; tile < nTiles;
       tile += stride) {
    long long n0 = tile * 16;
    v8f acc[4];
#pragma unroll
    for (int nt = 0; nt < 4; ++nt)
#pragma unroll
      for (int r = 0; r < 8; ++r)
        acc[nt][r] = xh[(n0 + r + 8 * half) * HID + nt * 16 + ln]; // residual

    const float* xr = xh + (n0 + ln) * HID;
    const float* ar = agg + (n0 + ln) * HID;
#pragma unroll
    for (int kc = 0; kc < 4; ++kc) {
      int k0 = kc * 32;
      v16h a;
#pragma unroll
      for (int j = 0; j < 8; ++j) {
        int k1 = k0 + 16 * half + j;
        int k2 = k0 + 16 * half + 8 + j;
        a[j]     = (_Float16)(k1 < HID ? xr[k1] : ar[k1 - HID]);
        a[8 + j] = (_Float16)(k2 < HID ? xr[k2] : ar[k2 - HID]);
      }
#pragma unroll
      for (int nt = 0; nt < 4; ++nt) acc[nt] = wmma_f16(a, b[kc][nt], acc[nt]);
    }
#pragma unroll
    for (int nt = 0; nt < 4; ++nt)
#pragma unroll
      for (int r = 0; r < 8; ++r)
        hout[(n0 + r + 8 * half) * HID + nt * 16 + ln] = acc[nt][r];
  }
}

// --------------------------- irregular graph ops ---------------------------
// All vectorized by 4 channels: one thread = (edge|node, 4 channels).

__global__ void deg_init_kernel(float* __restrict__ deg) {
  long long t = (long long)blockIdx.x * 256 + threadIdx.x;
  if (t < (long long)N_NODES * (HID / 4))
    reinterpret_cast<float4*>(deg)[t] = make_float4(1.f, 1.f, 1.f, 1.f);
}

__global__ void deg_scatter_kernel(const int* __restrict__ ei,
                                   const float* __restrict__ ew,
                                   float* __restrict__ deg) {
  long long t = (long long)blockIdx.x * 256 + threadIdx.x;
  if (t >= (long long)N_EDGES * (HID / 4)) return;
  int cq = (int)(t & 15);
  long long e = t >> 4;
  int col = ei[N_EDGES + e];
  float4 w = *reinterpret_cast<const float4*>(ew + e * HID + cq * 4);
  float* dst = deg + (long long)col * HID + cq * 4;
  unsafeAtomicAdd(dst + 0, w.x);
  unsafeAtomicAdd(dst + 1, w.y);
  unsafeAtomicAdd(dst + 2, w.z);
  unsafeAtomicAdd(dst + 3, w.w);
}

__global__ void rsqrt_kernel(float* __restrict__ deg) {
  long long t = (long long)blockIdx.x * 256 + threadIdx.x;
  if (t >= (long long)N_NODES * (HID / 4)) return;
  float4 d = reinterpret_cast<float4*>(deg)[t];
  d.x = d.x > 0.f ? rsqrtf(d.x) : 0.f;
  d.y = d.y > 0.f ? rsqrtf(d.y) : 0.f;
  d.z = d.z > 0.f ? rsqrtf(d.z) : 0.f;
  d.w = d.w > 0.f ? rsqrtf(d.w) : 0.f;
  reinterpret_cast<float4*>(deg)[t] = d;
}

__global__ void norm_kernel(const int* __restrict__ ei,
                            const float* __restrict__ dinv,
                            float* __restrict__ ew) {
  long long t = (long long)blockIdx.x * 256 + threadIdx.x;
  if (t >= (long long)N_EDGES * (HID / 4)) return;
  int cq = (int)(t & 15);
  long long e = t >> 4;
  int row = ei[e], col = ei[N_EDGES + e];
  float4 w  = *reinterpret_cast<const float4*>(ew + e * HID + cq * 4);
  float4 dr = *reinterpret_cast<const float4*>(dinv + (long long)row * HID + cq * 4);
  float4 dc = *reinterpret_cast<const float4*>(dinv + (long long)col * HID + cq * 4);
  w.x *= dr.x * dc.x;
  w.y *= dr.y * dc.y;
  w.z *= dr.z * dc.z;
  w.w *= dr.w * dc.w;
  *reinterpret_cast<float4*>(ew + e * HID + cq * 4) = w;
}

// xw = relu(h)*w_conv ; agg seeded with bias + self-loop (norm = dinv^2, w=1)
__global__ void pre_kernel(const float* __restrict__ h,
                           const float* __restrict__ dinv,
                           const float* __restrict__ wconv,
                           const float* __restrict__ bconv,
                           float* __restrict__ xw, float* __restrict__ agg) {
  long long t = (long long)blockIdx.x * 256 + threadIdx.x;
  if (t >= (long long)N_NODES * (HID / 4)) return;
  int cq = (int)(t & 15);
  float4 hv = reinterpret_cast<const float4*>(h)[t];
  float4 wc = *reinterpret_cast<const float4*>(wconv + cq * 4);
  float4 bc = *reinterpret_cast<const float4*>(bconv + cq * 4);
  float4 di = reinterpret_cast<const float4*>(dinv)[t];
  float4 v;
  v.x = (hv.x > 0.f ? hv.x : 0.f) * wc.x;
  v.y = (hv.y > 0.f ? hv.y : 0.f) * wc.y;
  v.z = (hv.z > 0.f ? hv.z : 0.f) * wc.z;
  v.w = (hv.w > 0.f ? hv.w : 0.f) * wc.w;
  reinterpret_cast<float4*>(xw)[t] = v;
  float4 a;
  a.x = bc.x + di.x * di.x * v.x;
  a.y = bc.y + di.y * di.y * v.y;
  a.z = bc.z + di.z * di.z * v.z;
  a.w = bc.w + di.w * di.w * v.w;
  reinterpret_cast<float4*>(agg)[t] = a;
}

__global__ void edge_agg_kernel(const int* __restrict__ ei,
                                const float* __restrict__ norm,
                                const float* __restrict__ xw,
                                float* __restrict__ agg) {
  long long t = (long long)blockIdx.x * 256 + threadIdx.x;
  if (t >= (long long)N_EDGES * (HID / 4)) return;
  int cq = (int)(t & 15);
  long long e = t >> 4;
  int row = ei[e], col = ei[N_EDGES + e];
  float4 nm = *reinterpret_cast<const float4*>(norm + e * HID + cq * 4);
  float4 xv = *reinterpret_cast<const float4*>(xw + (long long)row * HID + cq * 4);
  float* dst = agg + (long long)col * HID + cq * 4;
  unsafeAtomicAdd(dst + 0, nm.x * xv.x);
  unsafeAtomicAdd(dst + 1, nm.y * xv.y);
  unsafeAtomicAdd(dst + 2, nm.z * xv.z);
  unsafeAtomicAdd(dst + 3, nm.w * xv.w);
}

// out = concat(x_, relu(h)) @ Wo^T  (OUT_DIM = 1)
__global__ void final_kernel(const float* __restrict__ xh,
                             const float* __restrict__ h,
                             const float* __restrict__ Wo,
                             float* __restrict__ out) {
  long long n = (long long)blockIdx.x * 256 + threadIdx.x;
  if (n >= N_NODES) return;
  const float4* xr = reinterpret_cast<const float4*>(xh + n * HID);
  const float4* hr = reinterpret_cast<const float4*>(h + n * HID);
  const float4* w0 = reinterpret_cast<const float4*>(Wo);
  const float4* w1 = reinterpret_cast<const float4*>(Wo + HID);
  float s = 0.f;
#pragma unroll
  for (int k = 0; k < HID / 4; ++k) {
    float4 a = xr[k], w = w0[k];
    s += a.x * w.x + a.y * w.y + a.z * w.z + a.w * w.w;
  }
#pragma unroll
  for (int k = 0; k < HID / 4; ++k) {
    float4 a = hr[k], w = w1[k];
    a.x = a.x > 0.f ? a.x : 0.f;
    a.y = a.y > 0.f ? a.y : 0.f;
    a.z = a.z > 0.f ? a.z : 0.f;
    a.w = a.w > 0.f ? a.w : 0.f;
    s += a.x * w.x + a.y * w.y + a.z * w.z + a.w * w.w;
  }
  out[n] = s;
}

// ---------------------------------------------------------------------------
extern "C" void kernel_launch(void* const* d_in, const int* in_sizes, int n_in,
                              void* d_out, int out_size, void* d_ws, size_t ws_size,
                              hipStream_t stream)
{
  const float* x         = (const float*)d_in[0];
  const int*   ei        = (const int*)d_in[1];
  const float* edge_attr = (const float*)d_in[2];
  const float* W1        = (const float*)d_in[3];
  const float* W2        = (const float*)d_in[4];
  const float* Wi        = (const float*)d_in[5];
  const float* bi        = (const float*)d_in[6];
  const float* w_conv    = (const float*)d_in[7];
  const float* b_conv    = (const float*)d_in[8];
  const float* Wl        = (const float*)d_in[9];
  const float* Wo        = (const float*)d_in[10];
  float* out = (float*)d_out;

  const size_t EH = (size_t)N_EDGES * HID;
  const size_t NH = (size_t)N_NODES * HID;
  float* ws      = (float*)d_ws;
  float* normbuf = ws;            // [E,64]  ew then norm (in place)
  float* dinv    = normbuf + EH;  // [N,64]  deg then dinv (in place)
  float* xh      = dinv + NH;     // [N,64]  x_
  float* hbuf    = xh + NH;       // [N,64]  h
  float* xw      = hbuf + NH;     // [N,64]
  float* agg     = xw + NH;       // [N,64]

  const int ehBlocks4 = (int)((EH / 4 + 255) / 256);   // 100000
  const int nhBlocks4 = (int)((NH / 4 + 255) / 256);   // 6250
  const int edgeGemmBlocks = 1024;                     // grid-stride, 8 waves ea.
  const int nodeGemmBlocks = 256;                      // grid-stride, 8 waves ea.

  // 1. edge weight MLP -> ew
  edge_mlp_kernel<<<edgeGemmBlocks, 256, 0, stream>>>(edge_attr, W1, W2, normbuf);
  // 2. GCN normalization
  deg_init_kernel<<<nhBlocks4, 256, 0, stream>>>(dinv);
  deg_scatter_kernel<<<ehBlocks4, 256, 0, stream>>>(ei, normbuf, dinv);
  rsqrt_kernel<<<nhBlocks4, 256, 0, stream>>>(dinv);
  norm_kernel<<<ehBlocks4, 256, 0, stream>>>(ei, dinv, normbuf);
  // 3. i_lin
  ilin_kernel<<<nodeGemmBlocks, 256, 0, stream>>>(x, Wi, bi, xh);
  // 4. layers
  const float* hcur = xh;
  for (int l = 0; l < N_LAYERS; ++l) {
    pre_kernel<<<nhBlocks4, 256, 0, stream>>>(hcur, dinv, w_conv + l * HID,
                                              b_conv + l * HID, xw, agg);
    edge_agg_kernel<<<ehBlocks4, 256, 0, stream>>>(ei, normbuf, xw, agg);
    mix_kernel<<<nodeGemmBlocks, 256, 0, stream>>>(xh, agg,
                                                   Wl + (size_t)l * HID * 2 * HID,
                                                   hbuf);
    hcur = hbuf;
  }
  // 5. output projection
  final_kernel<<<(N_NODES + 255) / 256, 256, 0, stream>>>(xh, hbuf, Wo, out);
}
